// All2AllCostVolume_52604759441833
// MI455X (gfx1250) — compile-verified, matrix operand order checked by reading, not run on hardware
//
#include <hip/hip_runtime.h>
#include <hip/hip_bf16.h>

// ---------------------------------------------------------------------------
// All2AllCostVolume for MI455X (gfx1250, wave32, WMMA 16x16x32 f16 + TDM)
// B=2, N=M=4096, C=64, K=16, KQ=32
// ---------------------------------------------------------------------------

typedef __attribute__((ext_vector_type(16))) _Float16 v16h;
typedef __attribute__((ext_vector_type(8)))  float    v8f;
typedef __attribute__((ext_vector_type(4)))  unsigned int v4u;
typedef __attribute__((ext_vector_type(8)))  int      v8i;
typedef __attribute__((ext_vector_type(4)))  int      v4i;

#define BN_SCALE 0.9999950000374997f /* 1/sqrt(1+1e-5) */
#define NB 2
#define NPTS 4096

// ---------------- WMMA fragment helpers (wave32, 16x16x32 f16) -------------
// A fragment: element (m,k); lane holds row m=lane&15, K split by half-wave:
//   v[0..7]  -> k = k0 + (lane>>4)*8 + i ; v[8..15] -> k = k0+16+(lane>>4)*8+i
__device__ __forceinline__ v16h fragA_h(const _Float16* base, int ld, int m0,
                                        int k0, int lane) {
  const int m = m0 + (lane & 15);
  const int h = (lane >> 4) * 8;
  const _Float16* p = base + (size_t)m * ld + k0;
  v16h a;
#pragma unroll
  for (int i = 0; i < 8; ++i) { a[i] = p[h + i]; a[8 + i] = p[16 + h + i]; }
  return a;
}

// B fragment from row-major f32 weights W[kdim][ncols] (global or LDS);
// element (k,n), lane holds column n=lane&15. Zero beyond kdim.
__device__ __forceinline__ v16h fragB_W(const float* W, int ncols, int kdim,
                                        int k0, int n0, int lane) {
  const int n = n0 + (lane & 15);
  const int h = (lane >> 4) * 8;
  v16h bfr;
#pragma unroll
  for (int i = 0; i < 8; ++i) {
    const int k1 = k0 + h + i;
    const int k2 = k0 + 16 + h + i;
    bfr[i]     = (k1 < kdim) ? (_Float16)W[(size_t)k1 * ncols + n] : (_Float16)0.0f;
    bfr[8 + i] = (k2 < kdim) ? (_Float16)W[(size_t)k2 * ncols + n] : (_Float16)0.0f;
  }
  return bfr;
}

// ---------------- TDM: DMA a 2D f32 weight matrix into LDS -----------------
// D# per CDNA5 ISA 8.3/8.4: group0 {count=1, lds_addr, global_addr, type=2},
// group1 {data_size=4B, tensor_dim0=ncols, tensor_dim1=kdim,
//         tile_dim0=ncols, tile_dim1=kdim, stride0=ncols}.
__device__ __forceinline__ void tdm_load_w(const float* gsrc, float* lds_dst,
                                           int kdim, int ncols) {
#if __has_builtin(__builtin_amdgcn_tensor_load_to_lds)
  const unsigned lds_addr =
      (unsigned)(unsigned long long)(void*)lds_dst;   // low 32b = LDS offset
  const unsigned long long ga = (unsigned long long)gsrc;
  v4u g0;
  g0[0] = 1u;                                          // count=1 (user D#)
  g0[1] = lds_addr;                                    // bits 63:32
  g0[2] = (unsigned)(ga & 0xffffffffu);                // global_addr lo
  g0[3] = (unsigned)((ga >> 32) & 0x01ffffffu) | (2u << 30); // addr hi | type=2
  const unsigned td0 = (unsigned)ncols, td1 = (unsigned)kdim;
  v8i g1;
  g1[0] = (int)(2u << 16);                             // data_size = 4B
  g1[1] = (int)((td0 & 0xffffu) << 16);                // tensor_dim0[15:0]
  g1[2] = (int)((td0 >> 16) | ((td1 & 0xffffu) << 16));// td0 hi | td1 lo
  g1[3] = (int)((td1 >> 16) | ((td0 & 0xffffu) << 16));// td1 hi | tile_dim0
  g1[4] = (int)(td1 & 0xffffu);                        // tile_dim1 (tile_dim2=0)
  g1[5] = (int)td0;                                    // dim0_stride[31:0]
  g1[6] = 0;
  g1[7] = 0;
  const v4i z4 = {0, 0, 0, 0};
#if defined(__clang_major__) && (__clang_major__ >= 23)
  const v8i z8 = {0, 0, 0, 0, 0, 0, 0, 0};
  __builtin_amdgcn_tensor_load_to_lds(g0, g1, z4, z4, z8, 0);
#else
  __builtin_amdgcn_tensor_load_to_lds(g0, g1, z4, z4, 0);
#endif
#else
  (void)gsrc; (void)lds_dst; (void)kdim; (void)ncols;
#endif
}

// Block-level weight staging: one wave issues the TDM load, waits TENSORcnt,
// barrier publishes LDS to the workgroup. Fallback: cooperative copy.
__device__ __forceinline__ void stage_weights(const float* g, float* lds,
                                              int kdim, int ncols) {
  __syncthreads();   // everyone done reading previous contents of lds
#if __has_builtin(__builtin_amdgcn_tensor_load_to_lds)
  if (threadIdx.x < 32) {
    tdm_load_w(g, lds, kdim, ncols);
    __builtin_amdgcn_s_wait_tensorcnt(0);
  }
#else
  for (int i = threadIdx.x; i < kdim * ncols; i += blockDim.x) lds[i] = g[i];
#endif
  __syncthreads();
}

// Generic fused GEMM layer for a 4-wave (128-thread) block:
//   OUT[rows x ncols] = act( BN_SCALE * (ACT[rows x kdim] @ W + bias) )
// ACT: LDS f16 row-major, lda, zero-padded to KC*32. Each wave owns column
// tiles: loads all KC B-fragments once, then sweeps the row tiles.
template <int KC>
static __device__ void gemm_layer(const _Float16* act, int lda, int rows,
                                  const float* W, const float* bias,
                                  int kdim, int ncols,
                                  _Float16* out, int ldo, bool relu) {
  const int lane = threadIdx.x & 31;
  const int wave = threadIdx.x >> 5;
  const int nw   = blockDim.x >> 5;
  const int rt = rows >> 4, ct = ncols >> 4;
  for (int c = wave; c < ct; c += nw) {
    const int n0 = c << 4;
    v16h bfr[KC];
#pragma unroll
    for (int kc = 0; kc < KC; ++kc)
      bfr[kc] = fragB_W(W, ncols, kdim, kc * 32, n0, lane);
    for (int rtile = 0; rtile < rt; ++rtile) {
      const int r0 = rtile << 4;
      v8f acc = {};
#pragma unroll
      for (int kc = 0; kc < KC; ++kc) {
        v16h a = fragA_h(act, lda, r0, kc * 32, lane);
        acc = __builtin_amdgcn_wmma_f32_16x16x32_f16(false, a, false, bfr[kc],
                                                     (short)0, acc, false, false);
      }
      // D layout: v[r] -> row r0 + r + 8*(lane>>4), col n0 + (lane&15)
      const int n  = n0 + (lane & 15);
      const int mb = r0 + ((lane >> 4) << 3);
      const float bv = bias ? bias[n] : 0.0f;
#pragma unroll
      for (int r = 0; r < 8; ++r) {
        float v = (acc[r] + bv) * BN_SCALE;
        if (relu) v = fmaxf(v, 0.0f);
        out[(size_t)(mb + r) * ldo + n] = (_Float16)v;
      }
    }
  }
}

// Ordered-uint mapping so float max can use atomicMax(unsigned).
__device__ __forceinline__ unsigned fmap(float f) {
  unsigned u = __float_as_uint(f);
  return (u & 0x80000000u) ? ~u : (u | 0x80000000u);
}
__device__ __forceinline__ float funmap(unsigned k) {
  return (k & 0x80000000u) ? __uint_as_float(k & 0x7fffffffu)
                           : __uint_as_float(~k);
}

// ---------------- K1: brute-force KNN (ascending-distance top-k) -----------
template <int KK>
__global__ void knn_kernel(const float* __restrict__ q_xyz,
                           const float* __restrict__ s_xyz,
                           int* __restrict__ idx_out, int Nq, int Ns) {
  const int b  = blockIdx.y;
  const int qi = blockIdx.x * blockDim.x + threadIdx.x;
  __shared__ float sx[256], sy[256], sz[256];
  float qx = 0.f, qy = 0.f, qz = 0.f;
  if (qi < Nq) {
    const float* q = q_xyz + ((size_t)b * Nq + qi) * 3;
    qx = q[0]; qy = q[1]; qz = q[2];
  }
  float bd[KK]; int bi[KK];
#pragma unroll
  for (int u = 0; u < KK; ++u) { bd[u] = 3.4e38f; bi[u] = 0; }
  float worst = 3.4e38f; int wslot = 0;
  for (int base = 0; base < Ns; base += 256) {
    const int j = base + (int)threadIdx.x;
    if (j < Ns) {
      const float* s = s_xyz + ((size_t)b * Ns + j) * 3;
      sx[threadIdx.x] = s[0]; sy[threadIdx.x] = s[1]; sz[threadIdx.x] = s[2];
    }
    __syncthreads();
    const int lim = min(256, Ns - base);
    for (int t = 0; t < lim; ++t) {
      const float dx = sx[t] - qx, dy = sy[t] - qy, dz = sz[t] - qz;
      const float d = dx * dx + dy * dy + dz * dz;
      if (d < worst) {
        bd[wslot] = d; bi[wslot] = base + t;
        worst = bd[0]; wslot = 0;
#pragma unroll
        for (int u = 1; u < KK; ++u)
          if (bd[u] > worst) { worst = bd[u]; wslot = u; }
      }
    }
    __syncthreads();
  }
  if (qi < Nq) {
    for (int u = 0; u < KK; ++u) {            // selection sort, ascending
      int best = u;
      for (int v = u + 1; v < KK; ++v) if (bd[v] < bd[best]) best = v;
      float td = bd[u]; bd[u] = bd[best]; bd[best] = td;
      int   ti = bi[u]; bi[u] = bi[best]; bi[best] = ti;
      idx_out[((size_t)b * Nq + qi) * KK + u] = bi[u];
    }
  }
}

// ---------------- K2: f16 cast + L2 norms of a [rows,64] descriptor --------
__global__ void prep_kernel(const float* __restrict__ src,
                            _Float16* __restrict__ dst_h,
                            float* __restrict__ norms, int rows) {
  const int r = blockIdx.x * blockDim.x + threadIdx.x;
  if (r >= rows) return;
  const float* p = src + (size_t)r * 64;
  float s = 0.f;
  for (int c = 0; c < 64; ++c) {
    const float v = p[c];
    s += v * v;
    dst_h[(size_t)r * 64 + c] = (_Float16)v;
  }
  norms[r] = sqrtf(s);
}

// ---------------- K3: self-neighborhood attentive descriptor (convs_2) -----
// 4 points per block, K=16 -> 64 rows. h = [g_feat(64), rela(3), dist(1)] -> 68.
__global__ void nbr_desc_kernel(const float* __restrict__ xyz,
                                const float* __restrict__ feats,
                                const int* __restrict__ self_idx,
                                const float* __restrict__ W0,
                                const float* __restrict__ W1,
                                const float* __restrict__ W2,
                                float* __restrict__ desc_out) {
  const int b  = blockIdx.y;
  const int p0 = blockIdx.x * 4;
  __shared__ _Float16 bufA[64 * 96];
  __shared__ _Float16 bufB[64 * 96];
  __shared__ _Float16 gfeat[64 * 64];
  __shared__ float wbuf[68 * 64];          // TDM weight staging (f32)
  __shared__ int   ids[64];
  __shared__ float cxyz[4][3];
  __shared__ float rmx[64];
  __shared__ float wgt[64];
  const int tid = threadIdx.x, lane = tid & 31, wave = tid >> 5;

  if (tid < 12) cxyz[tid / 3][tid % 3] =
      xyz[((size_t)b * NPTS + p0 + tid / 3) * 3 + tid % 3];
  if (tid < 64) ids[tid] =
      self_idx[((size_t)b * NPTS + p0 + (tid >> 4)) * 16 + (tid & 15)];
  __syncthreads();

  for (int r = wave; r < 64; r += 4) {
    const int p = r >> 4;
    const int j = ids[r];
    const float* f = feats + ((size_t)b * NPTS + j) * 64;
    const _Float16 h0 = (_Float16)f[lane];
    const _Float16 h1 = (_Float16)f[lane + 32];
    bufA[r * 96 + lane] = h0;       bufA[r * 96 + lane + 32] = h1;
    gfeat[r * 64 + lane] = h0;      gfeat[r * 64 + lane + 32] = h1;
    if (lane == 0) {
      const float* gx = xyz + ((size_t)b * NPTS + j) * 3;
      const float rx = gx[0] - cxyz[p][0];
      const float ry = gx[1] - cxyz[p][1];
      const float rz = gx[2] - cxyz[p][2];
      const float dd = sqrtf(rx * rx + ry * ry + rz * rz + 1e-20f);
      bufA[r * 96 + 64] = (_Float16)rx;
      bufA[r * 96 + 65] = (_Float16)ry;
      bufA[r * 96 + 66] = (_Float16)rz;
      bufA[r * 96 + 67] = (_Float16)dd;
      for (int c = 68; c < 96; ++c) bufA[r * 96 + c] = (_Float16)0.f;
    }
  }
  stage_weights(W0, wbuf, 68, 64);
  gemm_layer<3>(bufA, 96, 64, wbuf, nullptr, 68, 64, bufB, 96, true);
  stage_weights(W1, wbuf, 64, 64);
  gemm_layer<2>(bufB, 96, 64, wbuf, nullptr, 64, 64, bufA, 96, true);
  stage_weights(W2, wbuf, 64, 64);
  gemm_layer<2>(bufA, 96, 64, wbuf, nullptr, 64, 64, bufB, 96, true);
  __syncthreads();

  if (tid < 64) {                        // channel max per row
    float m = -3.4e38f;
    for (int c = 0; c < 64; ++c) m = fmaxf(m, (float)bufB[tid * 96 + c]);
    rmx[tid] = m;
  }
  __syncthreads();
  if (tid < 4) {                         // softmax over K=16 per point
    float m = -3.4e38f;
    for (int k = 0; k < 16; ++k) m = fmaxf(m, rmx[tid * 16 + k]);
    float s = 0.f;
    for (int k = 0; k < 16; ++k) {
      const float e = expf(rmx[tid * 16 + k] - m);
      wgt[tid * 16 + k] = e; s += e;
    }
    const float inv = 1.f / s;
    for (int k = 0; k < 16; ++k) wgt[tid * 16 + k] *= inv;
  }
  __syncthreads();
  for (int o = tid; o < 256; o += 128) { // weighted sum of g_feat
    const int p = o >> 6, c = o & 63;
    float s = 0.f;
    for (int k = 0; k < 16; ++k)
      s += wgt[p * 16 + k] * (float)gfeat[(p * 16 + k) * 64 + c];
    desc_out[((size_t)b * NPTS + p0 + p) * 64 + c] = s;
  }
}

// ---------------- K4: streaming cosine row/col max (WMMA, no [M,N] store) --
__global__ void cosmax_kernel(const _Float16* __restrict__ dm,   // [B,Mr,64]
                              const _Float16* __restrict__ dn,   // [B,Nc,64]
                              const float* __restrict__ nmv,
                              const float* __restrict__ nnv,
                              unsigned* __restrict__ rowmax_u,   // [B,Mr]
                              unsigned* __restrict__ colmax_u,   // [B,Nc]
                              int Mr, int Nc) {
  const int b    = blockIdx.y;
  const int m0   = blockIdx.x * 16;
  const int lane = threadIdx.x & 31, wave = threadIdx.x >> 5,
            nw   = blockDim.x >> 5;
  const _Float16* dmb = dm + (size_t)b * Mr * 64;
  const _Float16* dnb = dn + (size_t)b * Nc * 64;
  const v16h a0 = fragA_h(dmb, 64, m0, 0,  lane);
  const v16h a1 = fragA_h(dmb, 64, m0, 32, lane);
  const int mb = m0 + ((lane >> 4) << 3);
  float nmr[8], rm[8];
#pragma unroll
  for (int r = 0; r < 8; ++r) {
    nmr[r] = nmv[(size_t)b * Mr + mb + r];
    rm[r]  = -3.4e38f;
  }
  for (int n0 = wave * 16; n0 < Nc; n0 += nw * 16) {
    // B(k=c, n) = dn[n][c]: same striding pattern as an A fragment on row n
    const v16h b0 = fragA_h(dnb, 64, n0, 0,  lane);
    const v16h b1 = fragA_h(dnb, 64, n0, 32, lane);
    v8f acc = {};
    acc = __builtin_amdgcn_wmma_f32_16x16x32_f16(false, a0, false, b0,
                                                 (short)0, acc, false, false);
    acc = __builtin_amdgcn_wmma_f32_16x16x32_f16(false, a1, false, b1,
                                                 (short)0, acc, false, false);
    const int n = n0 + (lane & 15);
    const float nn = nnv[(size_t)b * Nc + n];
    float cm = -3.4e38f;
#pragma unroll
    for (int r = 0; r < 8; ++r) {
      const float cv = acc[r] / (nmr[r] * nn + 1e-6f);
      rm[r] = fmaxf(rm[r], cv);
      cm    = fmaxf(cm, cv);
    }
    atomicMax(&colmax_u[(size_t)b * Nc + n], fmap(cm));
  }
#pragma unroll
  for (int r = 0; r < 8; ++r)
    atomicMax(&rowmax_u[(size_t)b * Mr + mb + r], fmap(rm[r]));
}

// ---------------- K5: gathered bidirectional normalized cosine -------------
__global__ void gcos_kernel(const int* __restrict__ idx,          // [B,N,32]
                            const _Float16* __restrict__ dm,
                            const _Float16* __restrict__ dn,
                            const float* __restrict__ nmv,
                            const float* __restrict__ nnv,
                            const unsigned* __restrict__ rmu,
                            const unsigned* __restrict__ cmu,
                            float* __restrict__ d2s,
                            float* __restrict__ s2d, int Mr, int Nc) {
  const long long t = (long long)blockIdx.x * blockDim.x + threadIdx.x;
  const long long total = 2LL * Nc * 32;
  if (t >= total) return;
  const int n = (int)((t >> 5) % Nc);
  const int b = (int)(t / ((long long)Nc * 32));
  const int m = idx[t];
  const _Float16* pm = dm + ((size_t)b * Mr + m) * 64;
  const _Float16* pn = dn + ((size_t)b * Nc + n) * 64;
  float s = 0.f;
  for (int c = 0; c < 64; ++c) s += (float)pm[c] * (float)pn[c];
  const float cosv =
      s / (nmv[(size_t)b * Mr + m] * nnv[(size_t)b * Nc + n] + 1e-6f);
  d2s[t] = cosv / (funmap(rmu[(size_t)b * Mr + m]) + 1e-6f);
  s2d[t] = cosv / (funmap(cmu[(size_t)b * Nc + n]) + 1e-6f);
}

// ---------------- K6: fused stage-1 (convs1 + pir + mlp1 + pi_enc + mlp2) --
// One block per (b,n); 128 threads; KQ=32 rows in LDS.
__global__ void stage1_kernel(
    const float* __restrict__ wxyz, const float* __restrict__ wpts,
    const float* __restrict__ fxyz, const float* __restrict__ fpts,
    const int* __restrict__ knn_idx,
    const float* __restrict__ d2s, const float* __restrict__ s2d,
    const float* __restrict__ d2sn, const float* __restrict__ s2dn,
    const float* cw0, const float* cw1, const float* cw2,
    const float* pirW, const float* pirB,
    const float* m1w0, const float* m1w1, const float* m1w2,
    const float* m1b0, const float* m1b1, const float* m1b2,
    const float* pieW, const float* pieB,
    const float* m2w0, const float* m2w1,
    const float* m2b0, const float* m2b1,
    float* __restrict__ pi_out) {
  const int b = blockIdx.y;
  const int n = blockIdx.x;
  __shared__ _Float16 bufX[32 * 224];
  __shared__ _Float16 bufY[32 * 128];
  __shared__ _Float16 bufZ[32 * 128];
  __shared__ _Float16 pix[32 * 32];
  __shared__ float wb1[64 * 64];           // TDM weight staging (f32)
  __shared__ int   ids[32];
  __shared__ float dkx[32][3];
  __shared__ float rel[32][3];
  __shared__ float rds[32];
  __shared__ float sxyz[3];
  __shared__ float hm[32];
  __shared__ float aw[32];
  const int tid = threadIdx.x, lane = tid & 31, wave = tid >> 5;
  const size_t bn = (size_t)b * NPTS + n;

  if (tid == 0) __builtin_prefetch(cw0, 0, 2);
  if (tid < 3)  sxyz[tid] = wxyz[bn * 3 + tid];
  if (tid < 32) ids[tid]  = knn_idx[bn * 32 + tid];
  __syncthreads();
  if (tid < 32) {
    const float* g = fxyz + ((size_t)b * NPTS + ids[tid]) * 3;
    const float gx = g[0], gy = g[1], gz = g[2];
    dkx[tid][0] = gx; dkx[tid][1] = gy; dkx[tid][2] = gz;
    const float rx = gx - sxyz[0], ry = gy - sxyz[1], rz = gz - sxyz[2];
    rel[tid][0] = rx; rel[tid][1] = ry; rel[tid][2] = rz;
    rds[tid] = sqrtf(rx * rx + ry * ry + rz * rz + 1e-20f);
  }
  __syncthreads();

  // ---- build convs1 feats [32 x 142] (pad to 160) + pi_xyz_concat ----
  const float* sdesc = wpts + bn * 64;
  for (int r = wave; r < 32; r += 4) {
    const float* dd = fpts + ((size_t)b * NPTS + ids[r]) * 64;
    for (int c = lane; c < 160; c += 32) {
      float v;
      if      (c < 3)   v = rel[r][c];
      else if (c == 3)  v = rds[r];
      else if (c < 7)   v = sxyz[c - 4];
      else if (c < 10)  v = dkx[r][c - 7];
      else if (c < 74)  v = sdesc[c - 10];
      else if (c < 138) v = dd[c - 74];
      else if (c == 138) v = d2s[bn * 32 + r];
      else if (c == 139) v = s2d[bn * 32 + r];
      else if (c == 140) v = d2sn[bn * 32 + r];
      else if (c == 141) v = s2dn[bn * 32 + r];
      else v = 0.f;
      bufX[r * 224 + c] = (_Float16)v;
    }
    {
      const int c = lane;      // pi_xyz_concat = [sxyz, dknn_xyz, rela, dist]
      float v;
      if      (c < 3)  v = sxyz[c];
      else if (c < 6)  v = dkx[r][c - 3];
      else if (c < 9)  v = rel[r][c - 6];
      else if (c == 9) v = rds[r];
      else v = 0.f;
      pix[r * 32 + c] = (_Float16)v;
    }
  }
  __syncthreads();

  // ---- convs1: 142->128->128->64 ----
  gemm_layer<5>(bufX, 224, 32, cw0, nullptr, 142, 128, bufY, 128, true);
  __syncthreads();
  gemm_layer<4>(bufY, 128, 32, cw1, nullptr, 128, 128, bufZ, 128, true);
  __syncthreads();
  gemm_layer<4>(bufZ, 128, 32, cw2, nullptr, 128, 64, bufY, 128, true);
  __syncthreads();

  // ---- aw = softmax over KQ of channel max ----
  if (tid < 32) {
    float m = -3.4e38f;
    for (int c = 0; c < 64; ++c) m = fmaxf(m, (float)bufY[tid * 128 + c]);
    hm[tid] = m;
  }
  __syncthreads();
  if (tid == 0) {
    float m = -3.4e38f;
    for (int r = 0; r < 32; ++r) m = fmaxf(m, hm[r]);
    float s = 0.f;
    for (int r = 0; r < 32; ++r) { const float e = expf(hm[r] - m); aw[r] = e; s += e; }
    const float inv = 1.f / s;
    for (int r = 0; r < 32; ++r) aw[r] *= inv;
  }
  __syncthreads();

  // ---- attentive = conv(dst_knn_desc * aw, pirW, pirB) ----
  for (int r = wave; r < 32; r += 4)
    for (int c = lane; c < 64; c += 32)
      bufZ[r * 128 + c] = (_Float16)((float)bufX[r * 224 + 74 + c] * aw[r]);
  stage_weights(pirW, wb1, 64, 64);
  gemm_layer<2>(bufZ, 128, 32, wb1, pirB, 64, 64, bufX + 138, 224, true);
  // rewrite channels 0..9 of bufX as pi_xyz_concat; zero tail (disjoint cols)
  for (int r = wave; r < 32; r += 4) {
    if (lane < 10) {
      const float v = (lane < 3) ? sxyz[lane]
                    : (lane < 6) ? dkx[r][lane - 3]
                    : (lane < 9) ? rel[r][lane - 6] : rds[r];
      bufX[r * 224 + lane] = (_Float16)v;
    }
    for (int c = 202 + lane; c < 224; c += 32) bufX[r * 224 + c] = (_Float16)0.f;
  }
  __syncthreads();

  // ---- mlp1: 202->128->64->64 (pi_feat -> bufY[:, :64]) ----
  gemm_layer<7>(bufX, 224, 32, m1w0, m1b0, 202, 128, bufY, 128, true);
  __syncthreads();
  gemm_layer<4>(bufY, 128, 32, m1w1, m1b1, 128, 64, bufZ, 128, true);
  stage_weights(m1w2, wb1, 64, 64);
  gemm_layer<2>(bufZ, 128, 32, wb1, m1b2, 64, 64, bufY, 128, true);

  // ---- g = [pi_enc(pix) | pi_feat] -> mlp2 128->128->64 ----
  stage_weights(pieW, wb1, 10, 64);
  gemm_layer<1>(pix, 32, 32, wb1, pieB, 10, 64, bufZ, 128, true);
  for (int r = wave; r < 32; r += 4)
    for (int c = lane; c < 64; c += 32)
      bufZ[r * 128 + 64 + c] = bufY[r * 128 + c];   // keep pi_feat alongside
  __syncthreads();
  gemm_layer<4>(bufZ, 128, 32, m2w0, m2b0, 128, 128, bufY, 128, true);
  __syncthreads();
  gemm_layer<4>(bufY, 128, 32, m2w1, m2b1, 128, 64, bufX, 224, true);
  __syncthreads();

  // ---- WQ = softmax over KQ per channel; pi_feat_new = sum WQ*pi_feat ----
  if (tid < 64) {
    float m = -3.4e38f;
    for (int r = 0; r < 32; ++r) m = fmaxf(m, (float)bufX[r * 224 + tid]);
    float s = 0.f, acc = 0.f;
    for (int r = 0; r < 32; ++r) {
      const float e = expf((float)bufX[r * 224 + tid] - m);
      s += e;
      acc += e * (float)bufZ[r * 128 + 64 + tid];
    }
    pi_out[bn * 64 + tid] = acc / s;
  }
}

// ---------------- K7: fused stage-2 (pc_enc + mlp2n + softmax-K sum) -------
__global__ void stage2_kernel(const float* __restrict__ wxyz,
                              const float* __restrict__ wpts,
                              const int* __restrict__ self_idx,
                              const float* __restrict__ pi_feat,
                              const float* pceW, const float* pceB,
                              const float* n2w0, const float* n2w1,
                              const float* n2b0, const float* n2b1,
                              float* __restrict__ out) {
  const int b = blockIdx.y;
  const int n = blockIdx.x;
  __shared__ _Float16 bufA[16 * 192];
  __shared__ _Float16 bufB[16 * 128];
  __shared__ _Float16 pix[16 * 32];
  __shared__ float wb2[128 * 64];          // TDM weight staging (f32)
  __shared__ float gkeep[16 * 64];
  __shared__ int   ids[16];
  __shared__ float sxyz[3];
  const int tid = threadIdx.x, lane = tid & 31, wave = tid >> 5;
  const size_t bn = (size_t)b * NPTS + n;

  if (tid == 0) __builtin_prefetch(n2w0, 0, 2);
  if (tid < 3)  sxyz[tid] = wxyz[bn * 3 + tid];
  if (tid < 16) ids[tid]  = self_idx[bn * 16 + tid];
  __syncthreads();

  const float* sdesc = wpts + bn * 64;
  for (int r = wave; r < 16; r += 4) {
    const int j = ids[r];
    const float* pf = pi_feat + ((size_t)b * NPTS + j) * 64;
    for (int c = lane; c < 64; c += 32) {
      const float v = pf[c];
      gkeep[r * 64 + c]        = v;
      bufA[r * 192 + 128 + c]  = (_Float16)v;        // grouped
      bufA[r * 192 + 64 + c]   = (_Float16)sdesc[c]; // desc expand
    }
    if (lane == 0) {
      const float* gx = wxyz + ((size_t)b * NPTS + j) * 3;
      const float g0 = gx[0], g1 = gx[1], g2 = gx[2];
      const float rx = g0 - sxyz[0], ry = g1 - sxyz[1], rz = g2 - sxyz[2];
      const float dd = sqrtf(rx * rx + ry * ry + rz * rz + 1e-20f);
      const float pv[10] = {sxyz[0], sxyz[1], sxyz[2], g0, g1, g2, rx, ry, rz, dd};
      for (int c = 0; c < 10; ++c)  pix[r * 32 + c] = (_Float16)pv[c];
      for (int c = 10; c < 32; ++c) pix[r * 32 + c] = (_Float16)0.f;
    }
  }
  stage_weights(pceW, wb2, 10, 64);
  gemm_layer<1>(pix, 32, 16, wb2, pceB, 10, 64, bufA, 192, true); // pc_enc
  __syncthreads();
  gemm_layer<6>(bufA, 192, 16, n2w0, n2b0, 192, 128, bufB, 128, true);
  stage_weights(n2w1, wb2, 128, 64);
  gemm_layer<4>(bufB, 128, 16, wb2, n2b1, 128, 64, bufA, 192, true);
  __syncthreads();

  if (tid < 64) {          // WP = softmax over K per channel; weighted sum
    float m = -3.4e38f;
    for (int r = 0; r < 16; ++r) m = fmaxf(m, (float)bufA[r * 192 + tid]);
    float s = 0.f, acc = 0.f;
    for (int r = 0; r < 16; ++r) {
      const float e = expf((float)bufA[r * 192 + tid] - m);
      s += e; acc += e * gkeep[r * 64 + tid];
    }
    out[bn * 64 + tid] = acc / s;
  }
}

// ---------------------------------------------------------------------------
extern "C" void kernel_launch(void* const* d_in, const int* in_sizes, int n_in,
                              void* d_out, int out_size, void* d_ws,
                              size_t ws_size, hipStream_t stream) {
  const float* wxyz = (const float*)d_in[0];
  const float* wpts = (const float*)d_in[1];
  const float* fxyz = (const float*)d_in[2];
  const float* fpts = (const float*)d_in[3];
  // params flattened in dict-insertion order:
  // 0-2 convs2_W | 3-5 convs1_W | 6 pir_W 7 pir_b | 8-10 mlp1_W 11-13 mlp1_b
  // 14 pi_enc_W 15 pi_enc_b | 16-17 mlp2_W 18-19 mlp2_b
  // 20 pc_enc_W 21 pc_enc_b | 22-23 mlp2n_W 24-25 mlp2n_b
  const float* P[26];
  for (int i = 0; i < 26; ++i) P[i] = (const float*)d_in[4 + i];

  // ---- workspace partition ----
  char* w = (char*)d_ws;
  size_t off = 0;
  auto alloc = [&](size_t bytes) -> void* {
    void* p = w + off;
    off += (bytes + 255) & ~(size_t)255;
    return p;
  };
  const size_t R = (size_t)NB * NPTS;
  int* src_knn  = (int*)alloc(R * 32 * 4);
  int* src_self = (int*)alloc(R * 16 * 4);
  int* dst_self = (int*)alloc(R * 16 * 4);
  _Float16* wp_h  = (_Float16*)alloc(R * 64 * 2);
  _Float16* fp_h  = (_Float16*)alloc(R * 64 * 2);
  float* wp_nm = (float*)alloc(R * 4);
  float* fp_nm = (float*)alloc(R * 4);
  float* src_nbr = (float*)alloc(R * 64 * 4);
  float* dst_nbr = (float*)alloc(R * 64 * 4);
  _Float16* snb_h = (_Float16*)alloc(R * 64 * 2);
  _Float16* dnb_h = (_Float16*)alloc(R * 64 * 2);
  float* snb_nm = (float*)alloc(R * 4);
  float* dnb_nm = (float*)alloc(R * 4);
  unsigned* rmax_p  = (unsigned*)alloc(R * 4);
  unsigned* cmax_p  = (unsigned*)alloc(R * 4);
  unsigned* rmax_nb = (unsigned*)alloc(R * 4);
  unsigned* cmax_nb = (unsigned*)alloc(R * 4);
  float* d2s  = (float*)alloc(R * 32 * 4);
  float* s2d  = (float*)alloc(R * 32 * 4);
  float* d2sn = (float*)alloc(R * 32 * 4);
  float* s2dn = (float*)alloc(R * 32 * 4);
  float* pi_feat = (float*)alloc(R * 64 * 4);
  (void)ws_size; (void)in_sizes; (void)n_in; (void)out_size;

  // ---- 1. KNN (cross KQ=32, self K=16 both clouds) ----
  dim3 kb(256), kg(NPTS / 256, NB);
  knn_kernel<32><<<kg, kb, 0, stream>>>(wxyz, fxyz, src_knn, NPTS, NPTS);
  knn_kernel<16><<<kg, kb, 0, stream>>>(wxyz, wxyz, src_self, NPTS, NPTS);
  knn_kernel<16><<<kg, kb, 0, stream>>>(fxyz, fxyz, dst_self, NPTS, NPTS);

  // ---- 2. f16 casts + norms of point descriptors ----
  prep_kernel<<<dim3((unsigned)(R / 256)), dim3(256), 0, stream>>>(wpts, wp_h, wp_nm, (int)R);
  prep_kernel<<<dim3((unsigned)(R / 256)), dim3(256), 0, stream>>>(fpts, fp_h, fp_nm, (int)R);

  // ---- 3. neighborhood attentive descriptors (convs_2) ----
  dim3 ng(NPTS / 4, NB), nb(128);
  nbr_desc_kernel<<<ng, nb, 0, stream>>>(wxyz, wpts, src_self, P[0], P[1], P[2], src_nbr);
  nbr_desc_kernel<<<ng, nb, 0, stream>>>(fxyz, fpts, dst_self, P[0], P[1], P[2], dst_nbr);
  prep_kernel<<<dim3((unsigned)(R / 256)), dim3(256), 0, stream>>>(src_nbr, snb_h, snb_nm, (int)R);
  prep_kernel<<<dim3((unsigned)(R / 256)), dim3(256), 0, stream>>>(dst_nbr, dnb_h, dnb_nm, (int)R);

  // ---- 4. streaming cosine max (rows: dst over n; cols: src over m) ----
  hipMemsetAsync(rmax_p,  0, R * 4, stream);
  hipMemsetAsync(cmax_p,  0, R * 4, stream);
  hipMemsetAsync(rmax_nb, 0, R * 4, stream);
  hipMemsetAsync(cmax_nb, 0, R * 4, stream);
  dim3 cg(NPTS / 16, NB), cb(128);
  cosmax_kernel<<<cg, cb, 0, stream>>>(fp_h,  wp_h,  fp_nm,  wp_nm,  rmax_p,  cmax_p,  NPTS, NPTS);
  cosmax_kernel<<<cg, cb, 0, stream>>>(dnb_h, snb_h, dnb_nm, snb_nm, rmax_nb, cmax_nb, NPTS, NPTS);

  // ---- 5. gathered bidirectional normalized cosine at cross-KNN sites ----
  const int gtot = NB * NPTS * 32;
  gcos_kernel<<<dim3(gtot / 256), dim3(256), 0, stream>>>(
      src_knn, fp_h, wp_h, fp_nm, wp_nm, rmax_p, cmax_p, d2s, s2d, NPTS, NPTS);
  gcos_kernel<<<dim3(gtot / 256), dim3(256), 0, stream>>>(
      src_knn, dnb_h, snb_h, dnb_nm, snb_nm, rmax_nb, cmax_nb, d2sn, s2dn, NPTS, NPTS);

  // ---- 6. fused stage 1 -> pi_feat_new [B,N,64] ----
  stage1_kernel<<<dim3(NPTS, NB), dim3(128), 0, stream>>>(
      wxyz, wpts, fxyz, fpts, src_knn, d2s, s2d, d2sn, s2dn,
      P[3], P[4], P[5], P[6], P[7],
      P[8], P[9], P[10], P[11], P[12], P[13],
      P[14], P[15], P[16], P[17], P[18], P[19], pi_feat);

  // ---- 7. fused stage 2 -> output [B,N,64] ----
  stage2_kernel<<<dim3(NPTS, NB), dim3(128), 0, stream>>>(
      wxyz, wpts, src_self, pi_feat,
      P[20], P[21], P[22], P[23], P[24], P[25], (float*)d_out);
}
